// TaskAlignedAssigner_53790170415245
// MI455X (gfx1250) — compile-verified, matrix operand order checked by reading, not run on hardware
//
#include <hip/hip_runtime.h>
#include <hip/hip_bf16.h>
#include <stdint.h>

#define NUM_ANCHORS 100000
#define NUM_CLASSES 80
#define NUM_GT      512
#define TOPK        13
#define EPSF        1e-9f
#define NTILES      (NUM_ANCHORS / 16)   // 6250, exact

typedef __attribute__((ext_vector_type(16))) _Float16     v16h;
typedef __attribute__((ext_vector_type(8)))  float        v8f;
typedef __attribute__((ext_vector_type(4)))  unsigned int v4u;
typedef __attribute__((ext_vector_type(8)))  int          v8i;
typedef __attribute__((ext_vector_type(4)))  int          v4i;

// K index held by element e (0..15) of a 16-bit A/B WMMA operand vector,
// for half-wave `hi` (lane>=16), per CDNA5 ISA 16-bit 16x32 layout:
//   lanes 0-15:  V0-V3 -> K 0..7,  V4-V7 -> K 16..23
//   lanes 16-31: V0-V3 -> K 8..15, V4-V7 -> K 24..31
__device__ __forceinline__ int kmap(int e, int hi) {
    int p = e >> 1, lo = e & 1;
    return hi * 8 + 2 * p + ((p >= 4) ? 8 : 0) + lo;
}

__global__ __launch_bounds__(256)
void taa_main_kernel(const float* __restrict__ pd_scores,
                     const float* __restrict__ pd_bboxes,
                     const int*   __restrict__ gt_labels,
                     const float* __restrict__ gt_bboxes,
                     float* __restrict__ out_labels,
                     float* __restrict__ out_bboxes,
                     float* __restrict__ out_maxmet,
                     float* __restrict__ out_metrics)
{
    __shared__ float4 sh_box[NUM_GT];   // 8 KB, filled by TDM
    __shared__ int    sh_lab[NUM_GT];   // 2 KB

    const int tid = threadIdx.x;

    for (int i = tid; i < NUM_GT; i += 256) sh_lab[i] = gt_labels[i];

    // --- Tensor Data Mover: DMA gt_bboxes (2048 f32) into LDS (wave 0 only) ---
    if ((tid >> 5) == 0) {
        uint64_t ga  = (uint64_t)(uintptr_t)gt_bboxes;
        uint32_t lds = (uint32_t)(uintptr_t)(&sh_box[0]);
        v4u g0;
        g0[0] = 1u;                                      // count=1 valid descriptor
        g0[1] = lds;                                     // lds_addr
        g0[2] = (uint32_t)(ga & 0xFFFFFFFFull);          // global_addr[31:0]
        g0[3] = (uint32_t)((ga >> 32) & 0x1FFFFFFull)    // global_addr[56:32]
              | 0x80000000u;                             // type=2 ("image")
        v8i g1;
        g1[0] = (int)(2u << 16);                         // data_size = 4 bytes
        g1[1] = (int)((2048u & 0xFFFFu) << 16);          // tensor_dim0 = 2048 (lo16)
        g1[2] = (int)(((2048u >> 16) & 0xFFFFu)          // tensor_dim0 hi16
              | (1u << 16));                             // tensor_dim1 = 1 (lo16)
        g1[3] = (int)(2048u << 16);                      // tile_dim0 = 2048
        g1[4] = 1;                                       // tile_dim1 = 1, tile_dim2 = 0
        g1[5] = 2048;                                    // tensor_dim0_stride
        g1[6] = 0;
        g1[7] = 0;
        v4i z4 = {0, 0, 0, 0};
        v8i z8 = {0, 0, 0, 0, 0, 0, 0, 0};
        __builtin_amdgcn_tensor_load_to_lds(g0, g1, z4, z4, z8, 0);
        __builtin_amdgcn_s_wait_tensorcnt(0);
    }
    __syncthreads();

    const int lane = tid & 31;
    const int wave = tid >> 5;
    const int n    = lane & 15;
    const int hi   = lane >> 4;
    const int tile = blockIdx.x * 8 + wave;
    if (tile >= NTILES) return;
    const int a0 = tile * 16;
    const int a  = a0 + n;                 // this lane's anchor column

    const float4 ab    = ((const float4*)pd_bboxes)[a];
    const float  areaA = (ab.z - ab.x) * (ab.w - ab.y);

    // B = pd_scores^T tile [K=96(padded) x N=16] in f16, 3 chunks of K=32.
    v16h bch[3];
#pragma unroll
    for (int c = 0; c < 3; ++c) {
#pragma unroll
        for (int e = 0; e < 16; ++e) {
            int k = c * 32 + kmap(e, hi);
            _Float16 v = (_Float16)0.0f;
            if (k < NUM_CLASSES) v = (_Float16)pd_scores[(size_t)a * NUM_CLASSES + k];
            bch[c][e] = v;
        }
    }

    float maxv = -1e30f;
    int   maxg = 0;

    for (int gt = 0; gt < NUM_GT / 16; ++gt) {
        const int g0   = gt * 16;
        const int labn = sh_lab[g0 + n];   // A-matrix row M = lane&15
        v8f acc = {};
#pragma unroll
        for (int c = 0; c < 3; ++c) {
            v16h av;                       // one-hot(label) row, exact in f16
#pragma unroll
            for (int e = 0; e < 16; ++e) {
                int k = c * 32 + kmap(e, hi);
                av[e] = (labn == k) ? (_Float16)1.0f : (_Float16)0.0f;
            }
            acc = __builtin_amdgcn_wmma_f32_16x16x32_f16(
                      false, av, false, bch[c], (short)0, acc, false, false);
        }
        // D layout: element r -> GT row g0 + hi*8 + r, anchor col = a
#pragma unroll
        for (int r = 0; r < 8; ++r) {
            const int    g  = g0 + hi * 8 + r;
            const float4 gb = sh_box[g];
            float ix1 = fmaxf(gb.x, ab.x), iy1 = fmaxf(gb.y, ab.y);
            float ix2 = fminf(gb.z, ab.z), iy2 = fminf(gb.w, ab.w);
            float iw = fmaxf(ix2 - ix1, 0.f), ih = fmaxf(iy2 - iy1, 0.f);
            float inter = iw * ih;
            float areaG = (gb.z - gb.x) * (gb.w - gb.y);
            float iou   = inter / (areaG + areaA - inter + EPSF);
            float i2    = iou * iou;
            float m     = acc[r] * (i2 * i2 * i2);     // score^1 * iou^6
            out_metrics[(size_t)g * NUM_ANCHORS + a] = m;
            if (m > maxv) { maxv = m; maxg = g; }
        }
    }

    // Merge the two GT-row halves held by lanes n and n+16 (wave32 shuffle).
    float ov = __shfl_xor(maxv, 16);
    int   og = __shfl_xor(maxg, 16);
    if (ov > maxv || (ov == maxv && og < maxg)) { maxv = ov; maxg = og; }

    if (hi == 0) {
        out_labels[a] = (maxv <= 0.f) ? (float)NUM_CLASSES : (float)sh_lab[maxg];
        float4 tb = sh_box[maxg];
        out_bboxes[a * 4 + 0] = tb.x;
        out_bboxes[a * 4 + 1] = tb.y;
        out_bboxes[a * 4 + 2] = tb.z;
        out_bboxes[a * 4 + 3] = tb.w;
        out_maxmet[a] = maxv;
    }
}

// One block per GT row: find the 13th-largest metric, rewrite row as 0/1 mask.
__global__ __launch_bounds__(256)
void taa_topk_kernel(float* __restrict__ metrics)
{
    __shared__ float cand[256 * TOPK];
    __shared__ float red_v[256];
    __shared__ int   red_i[256];
    __shared__ float winners[TOPK];

    const int g   = blockIdx.x;
    const int tid = threadIdx.x;
    float* row = metrics + (size_t)g * NUM_ANCHORS;

    float tk[TOPK];
#pragma unroll
    for (int j = 0; j < TOPK; ++j) tk[j] = -1e30f;

    for (int a = tid; a < NUM_ANCHORS; a += 256) {
        float m = row[a];
        if (m > tk[TOPK - 1]) {
            tk[TOPK - 1] = m;
#pragma unroll
            for (int j = TOPK - 1; j >= 1; --j) {
                if (tk[j] > tk[j - 1]) { float t = tk[j - 1]; tk[j - 1] = tk[j]; tk[j] = t; }
            }
        }
    }
#pragma unroll
    for (int j = 0; j < TOPK; ++j) cand[tid * TOPK + j] = tk[j];
    __syncthreads();

    for (int round = 0; round < TOPK; ++round) {
        float bv = -1e30f; int bi = tid * TOPK;
#pragma unroll
        for (int j = 0; j < TOPK; ++j) {
            float v = cand[tid * TOPK + j];
            if (v > bv) { bv = v; bi = tid * TOPK + j; }
        }
        red_v[tid] = bv; red_i[tid] = bi;
        __syncthreads();
        for (int s = 128; s > 0; s >>= 1) {
            if (tid < s) {
                float v2 = red_v[tid + s]; int i2 = red_i[tid + s];
                if (v2 > red_v[tid] || (v2 == red_v[tid] && i2 < red_i[tid])) {
                    red_v[tid] = v2; red_i[tid] = i2;
                }
            }
            __syncthreads();
        }
        if (tid == 0) { winners[round] = red_v[0]; cand[red_i[0]] = -1e30f; }
        __syncthreads();
    }

    const float thr = winners[TOPK - 1];
    for (int a = tid; a < NUM_ANCHORS; a += 256) {
        row[a] = (row[a] >= thr) ? 1.0f : 0.0f;
    }
}

extern "C" void kernel_launch(void* const* d_in, const int* in_sizes, int n_in,
                              void* d_out, int out_size, void* d_ws, size_t ws_size,
                              hipStream_t stream) {
    (void)in_sizes; (void)n_in; (void)out_size; (void)d_ws; (void)ws_size;
    const float* pd_scores = (const float*)d_in[0];
    const float* pd_bboxes = (const float*)d_in[1];
    const int*   gt_labels = (const int*)d_in[2];
    const float* gt_bboxes = (const float*)d_in[3];

    float* out         = (float*)d_out;
    float* out_labels  = out;                          // [A]
    float* out_bboxes  = out + NUM_ANCHORS;            // [A,4]
    float* out_maxmet  = out + (size_t)NUM_ANCHORS * 5;// [A]
    float* out_metrics = out + (size_t)NUM_ANCHORS * 6;// [G,A] metrics -> mask

    const int nblocks = (NTILES + 7) / 8;              // 8 waves/block, 1 tile/wave
    taa_main_kernel<<<nblocks, 256, 0, stream>>>(pd_scores, pd_bboxes, gt_labels,
                                                 gt_bboxes, out_labels, out_bboxes,
                                                 out_maxmet, out_metrics);
    taa_topk_kernel<<<NUM_GT, 256, 0, stream>>>(out_metrics);
}
